// ASSA_52888227283158
// MI455X (gfx1250) — compile-verified, matrix operand order checked by reading, not run on hardware
//
#include <hip/hip_runtime.h>

typedef __attribute__((ext_vector_type(16))) _Float16 v16h;
typedef __attribute__((ext_vector_type(8)))  _Float16 v8h;
typedef __attribute__((ext_vector_type(4)))  _Float16 v4h;
typedef __attribute__((ext_vector_type(8)))  float    v8f;

#define C_DIM    512
#define WIN      8
#define NTOK     64
#define HEADS    16
#define HEAD_DIM 32
#define HGRID    256
#define WGRID    256
#define WINDOWS  2048
#define QK_SCALE 0.17677669529663687f  /* 32^-0.5 */
#define LN_EPS   1e-5f

/* dynamic LDS layout (bytes) */
#define XS_OFF     0                       /* f16 [64][512]  = 64KB */
#define OUTS_OFF   65536                   /* f16 [64][512]  = 64KB */
#define WAVE_OFF   131072
#define WAVE_BYTES 18432
#define Q_OFF  0                           /* f16 [64][32] */
#define K_OFF  4096                        /* f16 [64][32] */
#define V_OFF  8192                        /* f16 [32][64] (transposed) */
#define S_OFF  12288                       /* f32 [16][64] */
#define A_OFF  16384                       /* f16 [16][64] */
#define SMEM_BYTES (WAVE_OFF + 8 * WAVE_BYTES)   /* 278528 */

/* ---- fragment loader -------------------------------------------------
 * 16-bit A-matrix 16x32 layout (ISA 7.12.2): lane l holds row (l&15);
 * element e maps to K = k0 + ((l>>4)<<3) + (e&7) + ((e>>3)<<4).
 * B fragments use the transposed-symmetric layout (lane holds column),
 * so loading from a [col][k] row-major array gives two contiguous
 * 16-byte reads per lane (ds_load_b128 / global_load_b128).
 */
__device__ __forceinline__ v16h frag_ld(const _Float16* base, int stride,
                                        int row0, int k0) {
  const int lane = threadIdx.x & 31;
  const _Float16* p = base + (size_t)(row0 + (lane & 15)) * stride
                           + (k0 + ((lane >> 4) << 3));
  v8h lo = *(const v8h*)p;         /* K = kb .. kb+7   */
  v8h hi = *(const v8h*)(p + 16);  /* K = kb+16..kb+23 */
  v16h r;
#pragma unroll
  for (int i = 0; i < 8; ++i) { r[i] = lo[i]; r[i + 8] = hi[i]; }
  return r;
}

__device__ __forceinline__ v8f wmma_f16(v16h a, v16h b, v8f c) {
  return __builtin_amdgcn_wmma_f32_16x16x32_f16(
      /*neg_a=*/false, a, /*neg_b=*/false, b,
      /*c_mod=*/(short)0, c, /*reuse_a=*/false, /*reuse_b=*/false);
}

/* ---- weight fp32 -> f16 transpose ([in,out] -> [out][in]) ---- */
__global__ void convert_weights(const float* __restrict__ wq,
                                const float* __restrict__ wkv,
                                const float* __restrict__ pw,
                                _Float16* __restrict__ wqT,
                                _Float16* __restrict__ wkvT,
                                _Float16* __restrict__ pT) {
  const int i = blockIdx.x * 256 + threadIdx.x;
  if (i < 262144) {                      /* wq: 512x512 */
    const int o = i >> 9, k = i & 511;
    wqT[i] = (_Float16)wq[k * 512 + o];
  } else if (i < 786432) {               /* wkv: 512x1024 */
    const int j = i - 262144;
    const int o = j >> 9, k = j & 511;
    wkvT[j] = (_Float16)wkv[k * 1024 + o];
  } else if (i < 1048576) {              /* proj_w: 512x512 */
    const int j = i - 786432;
    const int o = j >> 9, k = j & 511;
    pT[j] = (_Float16)pw[k * 512 + o];
  }
}

/* ---- fused: LN + window QKV + dual-branch attention + proj + residual ---- */
__global__ __launch_bounds__(256, 1)
void assa_window_attn(const float* __restrict__ x,
                      const float* __restrict__ n1g, const float* __restrict__ n1b,
                      const _Float16* __restrict__ wqT, const float* __restrict__ bq,
                      const _Float16* __restrict__ wkvT, const float* __restrict__ bkv,
                      const float* __restrict__ bias_table,
                      const float* __restrict__ w_mix,
                      const _Float16* __restrict__ projT, const float* __restrict__ proj_b,
                      float* __restrict__ out) {
  extern __shared__ char smem[];
  _Float16* Xs   = (_Float16*)(smem + XS_OFF);
  _Float16* Outs = (_Float16*)(smem + OUTS_OFF);
  const int wave = threadIdx.x >> 5;
  const int lane = threadIdx.x & 31;
  char* wbase = smem + WAVE_OFF + wave * WAVE_BYTES;
  _Float16* q_s  = (_Float16*)(wbase + Q_OFF);
  _Float16* k_s  = (_Float16*)(wbase + K_OFF);
  _Float16* vT_s = (_Float16*)(wbase + V_OFF);
  float*    s_s  = (float*)   (wbase + S_OFF);
  _Float16* a_s  = (_Float16*)(wbase + A_OFF);

  const int wid  = blockIdx.x;
  const int bidx = wid >> 10;
  const int wy   = (wid >> 5) & 31;
  const int wx   = wid & 31;

  /* softmax(w_mix) — 2 scalars, every block computes it */
  const float wm0 = w_mix[0], wm1 = w_mix[1];
  const float mw = fmaxf(wm0, wm1);
  const float e0 = __expf(wm0 - mw), e1 = __expf(wm1 - mw);
  const float w0 = e0 / (e0 + e1), w1 = e1 / (e0 + e1);

  /* ---- Stage A: LayerNorm the 64 window tokens into LDS (f16) ---- */
  {
    const int t  = threadIdx.x >> 2;      /* token 0..63, 4 threads each */
    const int q4 = threadIdx.x & 3;
    const int py = t >> 3, px = t & 7;
    const size_t tok = (size_t)bidx * (HGRID * WGRID)
                     + (size_t)(wy * WIN + py) * WGRID + (wx * WIN + px);
    const float* xp = x + tok * C_DIM + q4 * 128;
    float s1 = 0.f, s2 = 0.f;
#pragma unroll 8
    for (int j = 0; j < 128; j += 4) {
      float4 v = *(const float4*)(xp + j);
      s1 += v.x + v.y + v.z + v.w;
      s2 += v.x * v.x + v.y * v.y + v.z * v.z + v.w * v.w;
    }
    s1 += __shfl_xor(s1, 1); s1 += __shfl_xor(s1, 2);
    s2 += __shfl_xor(s2, 1); s2 += __shfl_xor(s2, 2);
    const float mean = s1 * (1.f / C_DIM);
    const float var  = s2 * (1.f / C_DIM) - mean * mean;
    const float rstd = rsqrtf(var + LN_EPS);
    _Float16* xr = Xs + t * C_DIM + q4 * 128;
#pragma unroll 8
    for (int j = 0; j < 128; j += 4) {
      float4 v  = *(const float4*)(xp + j);
      float4 g  = *(const float4*)(n1g + q4 * 128 + j);
      float4 bb = *(const float4*)(n1b + q4 * 128 + j);
      v4h h;
      h[0] = (_Float16)((v.x - mean) * rstd * g.x + bb.x);
      h[1] = (_Float16)((v.y - mean) * rstd * g.y + bb.y);
      h[2] = (_Float16)((v.z - mean) * rstd * g.z + bb.z);
      h[3] = (_Float16)((v.w - mean) * rstd * g.w + bb.w);
      *(v4h*)(xr + j) = h;
    }
  }
  __syncthreads();

  /* ---- Stage B: each wave processes heads (wave, wave+8) ---- */
  for (int hh = 0; hh < 2; ++hh) {
    const int h = wave + hh * 8;

    /* Q = Xs @ Wq[:, h*32:+32], scale+bias folded at store */
    for (int mt = 0; mt < 4; ++mt) {
#pragma unroll
      for (int nt = 0; nt < 2; ++nt) {
        v8f acc = {};
        for (int k0 = 0; k0 < C_DIM; k0 += 32) {
          v16h a = frag_ld(Xs, C_DIM, mt * 16, k0);
          v16h b = frag_ld(wqT, C_DIM, h * HEAD_DIM + nt * 16, k0);
          acc = wmma_f16(a, b, acc);
        }
        const int col  = nt * 16 + (lane & 15);
        const float bi = bq[h * HEAD_DIM + col];
#pragma unroll
        for (int j = 0; j < 8; ++j) {
          const int row = mt * 16 + j + ((lane >> 4) << 3);
          q_s[row * HEAD_DIM + col] = (_Float16)((acc[j] + bi) * QK_SCALE);
        }
      }
    }
    /* K and V (V stored transposed [d][token]) */
    for (int mt = 0; mt < 4; ++mt) {
#pragma unroll
      for (int nt = 0; nt < 2; ++nt) {
        v8f acck = {}, accv = {};
        for (int k0 = 0; k0 < C_DIM; k0 += 32) {
          v16h a  = frag_ld(Xs, C_DIM, mt * 16, k0);
          v16h bk = frag_ld(wkvT, C_DIM, h * HEAD_DIM + nt * 16, k0);
          v16h bv = frag_ld(wkvT, C_DIM, C_DIM + h * HEAD_DIM + nt * 16, k0);
          acck = wmma_f16(a, bk, acck);
          accv = wmma_f16(a, bv, accv);
        }
        const int col = nt * 16 + (lane & 15);
        const float bk0 = bkv[h * HEAD_DIM + col];
        const float bv0 = bkv[C_DIM + h * HEAD_DIM + col];
#pragma unroll
        for (int j = 0; j < 8; ++j) {
          const int row = mt * 16 + j + ((lane >> 4) << 3);
          k_s[row * HEAD_DIM + col] = (_Float16)(acck[j] + bk0);
          vT_s[col * NTOK + row]    = (_Float16)(accv[j] + bv0);
        }
      }
    }

    /* attention, 16 query rows at a time */
    for (int mt = 0; mt < 4; ++mt) {
      v16h qa = frag_ld(q_s, HEAD_DIM, mt * 16, 0);   /* K = head_dim = 32 */
#pragma unroll
      for (int nt = 0; nt < 4; ++nt) {
        v16h kb = frag_ld(k_s, HEAD_DIM, nt * 16, 0);
        v8f s = {};
        s = wmma_f16(qa, kb, s);
        const int m  = nt * 16 + (lane & 15);         /* key index   */
        const int mi = m >> 3, mj = m & 7;
#pragma unroll
        for (int j = 0; j < 8; ++j) {
          const int jr = j + ((lane >> 4) << 3);
          const int n  = mt * 16 + jr;                /* query index */
          const int ni = n >> 3, nj = n & 7;
          const int idx = (ni - mi + 7) * 15 + (nj - mj + 7);
          s_s[jr * NTOK + m] = s[j] + bias_table[idx * HEADS + h];
        }
      }
      /* dual-branch mix: softmax*w0 + relu(s)^2*w1 (wave-private LDS) */
      {
        const int r  = lane & 15;
        const int c0 = (lane >> 4) * 32;
        float mx = -3.4e38f;
        for (int c = 0; c < 32; ++c) mx = fmaxf(mx, s_s[r * NTOK + c0 + c]);
        mx = fmaxf(mx, __shfl_xor(mx, 16));
        float sum = 0.f;
        for (int c = 0; c < 32; ++c) sum += __expf(s_s[r * NTOK + c0 + c] - mx);
        sum += __shfl_xor(sum, 16);
        const float inv = 1.f / sum;
        for (int c = 0; c < 32; ++c) {
          const float sv = s_s[r * NTOK + c0 + c];
          const float sm = __expf(sv - mx) * inv;
          const float rl = fmaxf(sv, 0.f);
          a_s[r * NTOK + c0 + c] = (_Float16)(w0 * sm + w1 * rl * rl);
        }
      }
      /* out strip = attn[16,64] @ V[64,32] */
#pragma unroll
      for (int nt = 0; nt < 2; ++nt) {
        v8f acc = {};
#pragma unroll
        for (int kk = 0; kk < 2; ++kk) {
          v16h a = frag_ld(a_s, NTOK, 0, kk * 32);
          v16h b = frag_ld(vT_s, NTOK, nt * 16, kk * 32);
          acc = wmma_f16(a, b, acc);
        }
        const int col = h * HEAD_DIM + nt * 16 + (lane & 15);
#pragma unroll
        for (int j = 0; j < 8; ++j) {
          const int row = mt * 16 + j + ((lane >> 4) << 3);
          Outs[row * C_DIM + col] = (_Float16)acc[j];
        }
      }
    }
  }
  __syncthreads();

  /* ---- Stage C: projection + bias + residual, wave owns 64 out cols ---- */
  for (int mt = 0; mt < 4; ++mt) {
    for (int nt = 0; nt < 4; ++nt) {
      const int ncol = wave * 64 + nt * 16;
      v8f acc = {};
      for (int k0 = 0; k0 < C_DIM; k0 += 32) {
        v16h a = frag_ld(Outs, C_DIM, mt * 16, k0);
        v16h b = frag_ld(projT, C_DIM, ncol, k0);
        acc = wmma_f16(a, b, acc);
      }
      const int col  = ncol + (lane & 15);
      const float pb = proj_b[col];
#pragma unroll
      for (int j = 0; j < 8; ++j) {
        const int row = mt * 16 + j + ((lane >> 4) << 3);
        const int py = row >> 3, px = row & 7;
        const size_t tok = (size_t)bidx * (HGRID * WGRID)
                         + (size_t)(wy * WIN + py) * WGRID + (wx * WIN + px);
        const size_t g = tok * C_DIM + col;
        out[g] = acc[j] + pb + x[g];
      }
    }
  }
}

extern "C" void kernel_launch(void* const* d_in, const int* in_sizes, int n_in,
                              void* d_out, int out_size, void* d_ws, size_t ws_size,
                              hipStream_t stream) {
  (void)in_sizes; (void)n_in; (void)out_size; (void)ws_size;
  const float* x   = (const float*)d_in[0];
  const float* g1  = (const float*)d_in[1];
  const float* b1  = (const float*)d_in[2];
  const float* wq  = (const float*)d_in[3];
  const float* bq  = (const float*)d_in[4];
  const float* wkv = (const float*)d_in[5];
  const float* bkv = (const float*)d_in[6];
  const float* bt  = (const float*)d_in[7];
  const float* wm  = (const float*)d_in[8];
  const float* pw  = (const float*)d_in[9];
  const float* pb  = (const float*)d_in[10];
  float* out = (float*)d_out;

  char* ws = (char*)d_ws;
  _Float16* wqT  = (_Float16*)(ws);                      /* 512KB */
  _Float16* wkvT = (_Float16*)(ws + 524288);             /* 1MB   */
  _Float16* pT   = (_Float16*)(ws + 524288 + 1048576);   /* 512KB */

  convert_weights<<<4096, 256, 0, stream>>>(wq, wkv, pw, wqT, wkvT, pT);

  (void)hipFuncSetAttribute((const void*)assa_window_attn,
                            hipFuncAttributeMaxDynamicSharedMemorySize,
                            SMEM_BYTES);
  assa_window_attn<<<WINDOWS, 256, SMEM_BYTES, stream>>>(
      x, g1, b1, wqT, bq, wkvT, bkv, bt, wm, pT, pb, out);
}
